// ReactomeGNN_6751688589916
// MI455X (gfx1250) — matrix operand order repaired and checked by reading, not compiled, not run on hardware
//
#include <hip/hip_runtime.h>
#include <hip/hip_bf16.h>
#include <cstdint>
#include <cstddef>

// ---------------- problem constants (fixed by the reference) ----------------
#define NGENES   9229
#define NMOD     3
#define PROJ_F   32
#define HID_F    64
#define BATCH_B  16
#define NEDGES   295328
#define ETOT     (NEDGES + NGENES)        // edges + self loops = 304557
#define ROWS     (BATCH_B * NGENES)       // 147664, divisible by 16
#define NTILES   (ROWS / 16)              // 9229

typedef __attribute__((ext_vector_type(16))) _Float16 v16h;
typedef __attribute__((ext_vector_type(8)))  float    v8f;

// ---------------- helpers ----------------
__global__ void k_zero(float* __restrict__ p, size_t n) {
  size_t i = (size_t)blockIdx.x * blockDim.x + threadIdx.x;
  if (i < n) p[i] = 0.0f;
}

// deg starts at 1.0 (self loop), then count incoming edges
__global__ void k_deg_init(float* __restrict__ deg) {
  int i = blockIdx.x * blockDim.x + threadIdx.x;
  if (i < NGENES) deg[i] = 1.0f;
}

__global__ void k_deg_count(const long long* __restrict__ dst, float* __restrict__ deg) {
  int i = blockIdx.x * blockDim.x + threadIdx.x;
  if (i < NEDGES) atomicAdd(&deg[(int)dst[i]], 1.0f);
}

__global__ void k_dinv(float* __restrict__ deg) {
  int i = blockIdx.x * blockDim.x + threadIdx.x;
  if (i < NGENES) deg[i] = rsqrtf(deg[i]);   // deg >= 1 always
}

// norm[e] = dinv[src]*dinv[dst] for real edges; dinv[n]^2 for self loops
__global__ void k_norm(const long long* __restrict__ src, const long long* __restrict__ dst,
                       const float* __restrict__ dinv, float* __restrict__ norm) {
  int i = blockIdx.x * blockDim.x + threadIdx.x;
  if (i < NEDGES) {
    norm[i] = dinv[(int)src[i]] * dinv[(int)dst[i]];
  } else if (i < ETOT) {
    float d = dinv[i - NEDGES];
    norm[i] = d * d;
  }
}

// fusion: h0[b,n,p] = sum_m x[b, n*3+m] * projW[m,p] + projB[p]
__global__ void k_fusion(const float* __restrict__ x, const float* __restrict__ pW,
                         const float* __restrict__ pb, float* __restrict__ h0) {
  int i = blockIdx.x * blockDim.x + threadIdx.x;
  if (i >= BATCH_B * NGENES * PROJ_F) return;
  int p  = i & (PROJ_F - 1);
  int bn = i >> 5;
  const float* xr = x + (size_t)bn * NMOD;
  float acc = pb[p];
  acc += xr[0] * pW[0 * PROJ_F + p];
  acc += xr[1] * pW[1 * PROJ_F + p];
  acc += xr[2] * pW[2 * PROJ_F + p];
  h0[i] = acc;
}

// ---------------- WMMA GEMM: Out[rows,64] = A[rows,K] @ W[K,64] ----------------
// One wave per 16-row tile, 4 column tiles of 16, K stepped by 32 via
// v_wmma_f32_16x16x32_f16 (f16 inputs, f32 accumulate).
// Fragment layouts per CDNA5 ISA 7.12.2:
//  A 16x32 f16 : lane half hs selects K+8 offset; VGPR v<4 -> K=2v, v>=4 -> K=16+2(v-4)
//  B 32x16 f16 : col = lane%16; same K mapping as A (lane half hs selects K+8 offset; VGPR v<4 -> K=2v, v>=4 -> K=16+2(v-4))
//  C/D 16x16 f32: col = lane%16; VGPR r -> row r (lanes 0-15) / row r+8 (lanes 16-31)
template <int K>
__global__ void k_gemm_wmma(const float* __restrict__ A, const float* __restrict__ W,
                            float* __restrict__ Out) {
  const int lane = threadIdx.x & 31;
  const int wave = threadIdx.x >> 5;
  const int tile = blockIdx.x * (blockDim.x >> 5) + wave;
  if (tile >= NTILES) return;                     // wave-uniform: EXEC stays all-1s

  const int    m    = lane & 15;
  const int    hs   = lane >> 4;                  // lane half select
  const size_t row0 = (size_t)tile * 16;

  v8f acc[4];
#pragma unroll
  for (int nt = 0; nt < 4; ++nt) acc[nt] = (v8f){0.f,0.f,0.f,0.f,0.f,0.f,0.f,0.f};

  const float* arow = A + (row0 + (size_t)m) * K;

#pragma unroll
  for (int k0 = 0; k0 < K; k0 += 32) {
    // A fragment (convert f32 -> f16 on the fly)
    v16h afrag;
#pragma unroll
    for (int v = 0; v < 8; ++v) {
      int kb = k0 + ((v < 4) ? (2 * v) : (16 + 2 * (v - 4))) + 8 * hs;
      afrag[2 * v]     = (_Float16)arow[kb];
      afrag[2 * v + 1] = (_Float16)arow[kb + 1];
    }
#pragma unroll
    for (int nt = 0; nt < 4; ++nt) {
      const int col = nt * 16 + m;
      v16h bfrag;
#pragma unroll
      for (int v = 0; v < 8; ++v) {
        int kk = k0 + ((v < 4) ? (2 * v) : (16 + 2 * (v - 4))) + 8 * hs;
        bfrag[2 * v]     = (_Float16)W[(size_t)kk * HID_F + col];
        bfrag[2 * v + 1] = (_Float16)W[(size_t)(kk + 1) * HID_F + col];
      }
      acc[nt] = __builtin_amdgcn_wmma_f32_16x16x32_f16(
          /*neg_a=*/false, afrag, /*neg_b=*/false, bfrag,
          /*c_mod=*/(short)0, acc[nt], /*reuse_a=*/false, /*reuse_b=*/false);
    }
  }

#pragma unroll
  for (int nt = 0; nt < 4; ++nt) {
#pragma unroll
    for (int r = 0; r < 8; ++r) {
      Out[(row0 + (size_t)(r + 8 * hs)) * HID_F + nt * 16 + m] = acc[nt][r];
    }
  }
}

// ---------------- edge scatter: out[b,dst,f] += norm[e] * hw[b,src,f] ----------------
// Feature dim across consecutive threads -> coalesced 256B gather + coalesced f32 atomics,
// all L2-resident (38MB activations << 192MB L2).
__global__ void k_scatter(const float* __restrict__ hw, const long long* __restrict__ srcE,
                          const long long* __restrict__ dstE, const float* __restrict__ norm,
                          float* __restrict__ out) {
  long long tid = (long long)blockIdx.x * blockDim.x + threadIdx.x;
  const long long total = (long long)ETOT * BATCH_B * HID_F;
  if (tid >= total) return;
  int       f  = (int)(tid & (HID_F - 1));
  long long eb = tid >> 6;
  int       b  = (int)(eb & (BATCH_B - 1));
  int       e  = (int)(eb >> 4);
  int s, d;
  if (e < NEDGES) { s = (int)srcE[e]; d = (int)dstE[e]; }
  else            { s = d = e - NEDGES; }
  float v = norm[e] * hw[((size_t)b * NGENES + s) * HID_F + f];
  atomicAdd(&out[((size_t)b * NGENES + d) * HID_F + f], v);
}

__global__ void k_bias_relu(const float* __restrict__ in, const float* __restrict__ bias,
                            float* __restrict__ out, size_t total) {
  size_t i = (size_t)blockIdx.x * blockDim.x + threadIdx.x;
  if (i >= total) return;
  out[i] = fmaxf(in[i] + bias[i & (HID_F - 1)], 0.0f);
}

// pooled[b,f] = mean_n h[b,n,f]
__global__ void k_pool(const float* __restrict__ h, float* __restrict__ pooled) {
  int i = blockIdx.x * blockDim.x + threadIdx.x;
  if (i >= BATCH_B * HID_F) return;
  int f = i & (HID_F - 1);
  int b = i >> 6;
  const float* p = h + (size_t)b * NGENES * HID_F + f;
  float s = 0.0f;
  for (int n = 0; n < NGENES; ++n) s += p[(size_t)n * HID_F];
  pooled[i] = s * (1.0f / (float)NGENES);
}

__global__ void k_cls(const float* __restrict__ pooled, const float* __restrict__ W,
                      const float* __restrict__ bias, float* __restrict__ out) {
  int b = threadIdx.x;
  if (b >= BATCH_B) return;
  float s = bias[0];
#pragma unroll
  for (int f = 0; f < HID_F; ++f) s += pooled[b * HID_F + f] * W[f];
  out[b] = s;
}

// ---------------- launcher ----------------
extern "C" void kernel_launch(void* const* d_in, const int* in_sizes, int n_in,
                              void* d_out, int out_size, void* d_ws, size_t ws_size,
                              hipStream_t stream) {
  (void)in_sizes; (void)n_in; (void)out_size; (void)ws_size;
  const float*     x     = (const float*)d_in[0];
  const float*     projW = (const float*)d_in[1];
  const float*     projB = (const float*)d_in[2];
  const float*     W1    = (const float*)d_in[3];
  const float*     b1    = (const float*)d_in[4];
  const float*     W2    = (const float*)d_in[5];
  const float*     b2    = (const float*)d_in[6];
  const float*     clsW  = (const float*)d_in[7];
  const float*     clsB  = (const float*)d_in[8];
  const long long* ei    = (const long long*)d_in[9];   // int64 [2, NEDGES]
  const long long* srcE  = ei;
  const long long* dstE  = ei + NEDGES;
  float*           out   = (float*)d_out;

  // workspace carve-up (256B aligned regions)
  char* w = (char*)d_ws;
  auto take = [&](size_t bytes) -> char* {
    char* p = w;
    w += (bytes + (size_t)255) & ~(size_t)255;
    return p;
  };
  const size_t act_elems = (size_t)BATCH_B * NGENES * HID_F;   // 9,450,496
  float* deg    = (float*)take(sizeof(float) * NGENES);
  float* norm   = (float*)take(sizeof(float) * ETOT);
  float* pooled = (float*)take(sizeof(float) * BATCH_B * HID_F);
  float* buf0   = (float*)take(sizeof(float) * act_elems);     // activations
  float* buf1   = (float*)take(sizeof(float) * act_elems);     // GEMM out (hw)
  float* buf2   = (float*)take(sizeof(float) * act_elems);     // scatter accumulator

  const int       TB        = 256;
  const long long scat_tot  = (long long)ETOT * BATCH_B * HID_F;
  const int       scat_blks = (int)((scat_tot + TB - 1) / TB);
  const int       gemm_blks = (NTILES + 8 - 1) / 8;            // 8 waves / block

  // --- graph structure (recomputed every call; deterministic) ---
  k_deg_init<<<(NGENES + TB - 1) / TB, TB, 0, stream>>>(deg);
  k_deg_count<<<(NEDGES + TB - 1) / TB, TB, 0, stream>>>(dstE, deg);
  k_dinv<<<(NGENES + TB - 1) / TB, TB, 0, stream>>>(deg);
  k_norm<<<(ETOT + TB - 1) / TB, TB, 0, stream>>>(srcE, dstE, deg, norm);

  // --- fusion: x -> h0 [B,N,32] ---
  k_fusion<<<((BATCH_B * NGENES * PROJ_F) + TB - 1) / TB, TB, 0, stream>>>(x, projW, projB, buf0);

  // --- GCN layer 1 ---
  k_gemm_wmma<PROJ_F><<<gemm_blks, TB, 0, stream>>>(buf0, W1, buf1);
  k_zero<<<(int)((act_elems + TB - 1) / TB), TB, 0, stream>>>(buf2, act_elems);
  k_scatter<<<scat_blks, TB, 0, stream>>>(buf1, srcE, dstE, norm, buf2);
  k_bias_relu<<<(int)((act_elems + TB - 1) / TB), TB, 0, stream>>>(buf2, b1, buf0, act_elems);

  // --- GCN layer 2 ---
  k_gemm_wmma<HID_F><<<gemm_blks, TB, 0, stream>>>(buf0, W2, buf1);
  k_zero<<<(int)((act_elems + TB - 1) / TB), TB, 0, stream>>>(buf2, act_elems);
  k_scatter<<<scat_blks, TB, 0, stream>>>(buf1, srcE, dstE, norm, buf2);
  k_bias_relu<<<(int)((act_elems + TB - 1) / TB), TB, 0, stream>>>(buf2, b2, buf0, act_elems);

  // --- pool + classifier ---
  k_pool<<<(BATCH_B * HID_F + TB - 1) / TB, TB, 0, stream>>>(buf0, pooled);
  k_cls<<<1, 32, 0, stream>>>(pooled, clsW, clsB, out);
}
